// MambaVision_22909355557352
// MI455X (gfx1250) — compile-verified, matrix operand order checked by reading, not run on hardware
//
#include <hip/hip_runtime.h>
#include <math.h>

// ============================================================================
// MI455X (gfx1250, wave32) MambaVision forward.
// Core compute: WMMA everywhere it matters —
//   * k_gemm_wmma:    batched GEMM, 128x64 block tile, 8 waves, 4 back-to-back
//                     v_wmma_f32_16x16x32_f16 per k-step (A-fragment reuse),
//                     float4 (b128) fast-path staging, select-zero guards
//                     (no EXEC-divergent per-element load chains).
//   * k_conv3x3_wmma: implicit-GEMM conv (im2col gathered into LDS), same
//                     WMMA core, fused BN + activation + residual, NCHW out.
// VALU glue: layernorm, softmax, dwconv+SiLU, 8-state selective scan, etc.
// ============================================================================

typedef __attribute__((ext_vector_type(16))) _Float16 v16h;
typedef __attribute__((ext_vector_type(8)))  float    v8f;

#define ACT_NONE      0
#define ACT_RELU      1
#define ACT_GELU_TANH 2
#define ACT_GELU_ERF  3
#define ACT_SILU      4

__device__ __forceinline__ float apply_act(float x, int act) {
  switch (act) {
    case ACT_RELU:      return fmaxf(x, 0.0f);
    case ACT_GELU_TANH: {
      float t = tanhf(0.7978845608028654f * (x + 0.044715f * x * x * x));
      return 0.5f * x * (1.0f + t);
    }
    case ACT_GELU_ERF:  return 0.5f * x * (1.0f + erff(x * 0.7071067811865476f));
    case ACT_SILU:      return x / (1.0f + expf(-x));
    default:            return x;
  }
}

__device__ __forceinline__ float softplusf(float x) {
  return (x > 20.0f) ? x : log1pf(expf(x));
}

// Per-batch (blockIdx.z) pointer offset: off = (z/div)*s1 + (z%div)*s2
struct BatchMap { int div; long s1, s2; };

#define GEMM_BM 128
#define GEMM_BN 64
#define GEMM_BK 32

// Shared WMMA core helper: build A fragment for this wave from LDS.
// ISA 16-bit A layout: lanes 0-15 row M=lane, K 0-7 (v0..3) & 16-23 (v4..7);
// lanes 16-31 same rows, K 8-15 & 24-31.
#define BUILD_FRAG(frag, SRC, rowsel, kh)                                      \
  do {                                                                         \
    _Pragma("unroll")                                                          \
    for (int _i = 0; _i < 4; ++_i) {                                           \
      frag[2*_i]       = SRC[rowsel][kh + 2*_i];                               \
      frag[2*_i + 1]   = SRC[rowsel][kh + 2*_i + 1];                           \
      frag[8 + 2*_i]   = SRC[rowsel][kh + 16 + 2*_i];                          \
      frag[8 + 2*_i+1] = SRC[rowsel][kh + 16 + 2*_i + 1];                      \
    }                                                                          \
  } while (0)

// ---------------------------------------------------------------------------
// WMMA GEMM:  C[M,N] = act( alpha * A[M,K] x B' + bias[N] )
//   bT=1: B is [N,K] row-major (NT — weights);  bT=0: B is [K,N] (NN — att@V)
// ---------------------------------------------------------------------------
__global__ __launch_bounds__(256)
void k_gemm_wmma(const float* __restrict__ A, long lda,
                 const float* __restrict__ B, long ldb, int bT,
                 const float* __restrict__ bias,
                 float* __restrict__ C, long ldc,
                 int M, int N, int K, float alpha, int act,
                 BatchMap mA, BatchMap mB, BatchMap mC)
{
  __shared__ _Float16 As[GEMM_BM][GEMM_BK + 8];
  __shared__ _Float16 Bs[GEMM_BN][GEMM_BK + 8];

  int z = blockIdx.z;
  A += (long)(z / mA.div) * mA.s1 + (long)(z % mA.div) * mA.s2;
  B += (long)(z / mB.div) * mB.s1 + (long)(z % mB.div) * mB.s2;
  C += (long)(z / mC.div) * mC.s1 + (long)(z % mC.div) * mC.s2;

  const int m0  = blockIdx.y * GEMM_BM;
  const int n0  = blockIdx.x * GEMM_BN;
  const int tid = threadIdx.x;
  const int lane = tid & 31;
  const int wave = tid >> 5;

  v8f acc[4] = {};

  // Vectorized (b128) fast paths for interior tiles.
  const bool alA = ((lda & 3) == 0) && ((((unsigned long long)A) & 15ull) == 0);
  const bool alB = ((ldb & 3) == 0) && ((((unsigned long long)B) & 15ull) == 0);
  const bool fastA = alA && (m0 + GEMM_BM <= M);
  const bool fastB = alB && bT && (n0 + GEMM_BN <= N);

  for (int kk = 0; kk < K; kk += GEMM_BK) {
    const bool fullK = (kk + GEMM_BK <= K);
    // ---- stage A tile (128x32) ----
    if (fastA && fullK) {
      int r = tid >> 1, h = (tid & 1) * 16;
      const float4* p = reinterpret_cast<const float4*>(
          &A[(long)(m0 + r) * lda + kk + h]);
#pragma unroll
      for (int i = 0; i < 4; ++i) {
        float4 v = p[i];
        As[r][h + 4*i + 0] = (_Float16)v.x;
        As[r][h + 4*i + 1] = (_Float16)v.y;
        As[r][h + 4*i + 2] = (_Float16)v.z;
        As[r][h + 4*i + 3] = (_Float16)v.w;
      }
    } else {
#pragma unroll
      for (int t = tid; t < GEMM_BM * GEMM_BK; t += 256) {
        int r = t / GEMM_BK, c = t % GEMM_BK;
        int gm = m0 + r, gk = kk + c;
        bool ok = (gm < M) && (gk < K);
        float v = A[ok ? ((long)gm * lda + gk) : 0];   // unconditional load
        As[r][c] = ok ? (_Float16)v : (_Float16)0.0f;
      }
    }
    // ---- stage B tile (64x32) ----
    if (fastB && fullK) {
      int r = tid >> 2, q = (tid & 3) * 8;
      const float4* p = reinterpret_cast<const float4*>(
          &B[(long)(n0 + r) * ldb + kk + q]);
#pragma unroll
      for (int i = 0; i < 2; ++i) {
        float4 v = p[i];
        Bs[r][q + 4*i + 0] = (_Float16)v.x;
        Bs[r][q + 4*i + 1] = (_Float16)v.y;
        Bs[r][q + 4*i + 2] = (_Float16)v.z;
        Bs[r][q + 4*i + 3] = (_Float16)v.w;
      }
    } else {
#pragma unroll
      for (int t = tid; t < GEMM_BN * GEMM_BK; t += 256) {
        int r = t / GEMM_BK, c = t % GEMM_BK;
        int gn = n0 + r, gk = kk + c;
        bool ok = (gn < N) && (gk < K);
        long off = !ok ? 0 : (bT ? ((long)gn * ldb + gk)
                                 : ((long)gk * ldb + gn));
        float v = B[off];
        Bs[r][c] = ok ? (_Float16)v : (_Float16)0.0f;
      }
    }
    // prefetch next A K-tile (global_prefetch_b8)
    if (kk + GEMM_BK < K) {
      int gm = m0 + (tid % GEMM_BM);
      if (gm < M) __builtin_prefetch(&A[(long)gm * lda + kk + GEMM_BK], 0, 1);
    }
    __syncthreads();

    const int arow = wave * 16 + (lane & 15);
    const int kh   = (lane < 16) ? 0 : 8;
    v16h af;
    BUILD_FRAG(af, As, arow, kh);
#pragma unroll
    for (int tn = 0; tn < 4; ++tn) {
      const int bcol = tn * 16 + (lane & 15);
      v16h bf;
      BUILD_FRAG(bf, Bs, bcol, kh);
      acc[tn] = __builtin_amdgcn_wmma_f32_16x16x32_f16(
          false, af, false, bf, (short)0, acc[tn], false, false);
    }
    __syncthreads();
  }

  // ---- epilogue: D layout — VGPR i holds M = i (lanes 0-15) / 8+i (16-31) ----
  const int rbase = m0 + wave * 16 + ((lane < 16) ? 0 : 8);
#pragma unroll
  for (int tn = 0; tn < 4; ++tn) {
    int colg = n0 + tn * 16 + (lane & 15);
    float bsv = (bias && colg < N) ? bias[colg] : 0.0f;
#pragma unroll
    for (int i = 0; i < 8; ++i) {
      int rowg = rbase + i;
      if (rowg < M && colg < N) {
        float v = acc[tn][i] * alpha + bsv;
        C[(long)rowg * ldc + colg] = apply_act(v, act);
      }
    }
  }
}

// ---------------------------------------------------------------------------
// Implicit-GEMM 3x3 conv (pad 1): M = Nb*Ho*Wo, N = Co, K = Ci*9.
// A tile im2col-gathered into LDS; same 4-WMMA core; epilogue fuses
// inference BN + activation + residual and writes NCHW.
// ---------------------------------------------------------------------------
__global__ __launch_bounds__(256)
void k_conv3x3_wmma(const float* __restrict__ in, const float* __restrict__ w,
                    const float* bng, const float* bnb,
                    const float* bnm, const float* bnv, float eps,
                    const float* __restrict__ res, float* __restrict__ out,
                    int Nb, int Ci, int Hi, int Wi, int Co, int stride, int act)
{
  const int Ho = (Hi - 1) / stride + 1;
  const int Wo = (Wi - 1) / stride + 1;
  const int M = Nb * Ho * Wo, N = Co, K = Ci * 9;

  __shared__ _Float16 As[GEMM_BM][GEMM_BK + 8];
  __shared__ _Float16 Bs[GEMM_BN][GEMM_BK + 8];

  const int m0  = blockIdx.y * GEMM_BM;
  const int n0  = blockIdx.x * GEMM_BN;
  const int tid = threadIdx.x;
  const int lane = tid & 31;
  const int wave = tid >> 5;

  v8f acc[4] = {};

  for (int kk = 0; kk < K; kk += GEMM_BK) {
    // ---- A: on-the-fly im2col gather ----
#pragma unroll
    for (int t = tid; t < GEMM_BM * GEMM_BK; t += 256) {
      int r = t / GEMM_BK, c = t % GEMM_BK;
      int gm = m0 + r, gk = kk + c;
      bool ok = (gm < M) && (gk < K);
      int gmc = ok ? gm : 0, gkc = ok ? gk : 0;
      int wo = gmc % Wo; int t2 = gmc / Wo;
      int ho = t2 % Ho;  int nb = t2 / Ho;
      int ci = gkc / 9;  int t3 = gkc - ci * 9;
      int khh = t3 / 3, kww = t3 - khh * 3;
      int hi = ho * stride + khh - 1, wi = wo * stride + kww - 1;
      ok = ok && (hi >= 0) && (hi < Hi) && (wi >= 0) && (wi < Wi);
      long off = ok ? ((((long)nb * Ci + ci) * Hi + hi) * Wi + wi) : 0;
      float v = in[off];                               // unconditional load
      As[r][c] = ok ? (_Float16)v : (_Float16)0.0f;
    }
    // ---- B: weights [Co, Ci*9] (NT) ----
#pragma unroll
    for (int t = tid; t < GEMM_BN * GEMM_BK; t += 256) {
      int r = t / GEMM_BK, c = t % GEMM_BK;
      int gn = n0 + r, gk = kk + c;
      bool ok = (gn < N) && (gk < K);
      float v = w[ok ? ((long)gn * K + gk) : 0];
      Bs[r][c] = ok ? (_Float16)v : (_Float16)0.0f;
    }
    __syncthreads();

    const int arow = wave * 16 + (lane & 15);
    const int kh   = (lane < 16) ? 0 : 8;
    v16h af;
    BUILD_FRAG(af, As, arow, kh);
#pragma unroll
    for (int tn = 0; tn < 4; ++tn) {
      const int bcol = tn * 16 + (lane & 15);
      v16h bf;
      BUILD_FRAG(bf, Bs, bcol, kh);
      acc[tn] = __builtin_amdgcn_wmma_f32_16x16x32_f16(
          false, af, false, bf, (short)0, acc[tn], false, false);
    }
    __syncthreads();
  }

  // ---- epilogue: BN + act + residual, scatter to NCHW ----
  const int rbase = m0 + wave * 16 + ((lane < 16) ? 0 : 8);
#pragma unroll
  for (int tn = 0; tn < 4; ++tn) {
    int co = n0 + tn * 16 + (lane & 15);
    float scale = 1.0f, shift = 0.0f;
    if (bng && co < N) {
      float inv = rsqrtf(bnv[co] + eps) * bng[co];
      scale = inv; shift = bnb[co] - bnm[co] * inv;
    }
#pragma unroll
    for (int i = 0; i < 8; ++i) {
      int rowg = rbase + i;
      if (rowg < M && co < N) {
        int wo = rowg % Wo; int t2 = rowg / Wo;
        int ho = t2 % Ho;   int nb = t2 / Ho;
        long o = (((long)nb * Co + co) * Ho + ho) * Wo + wo;
        float v = apply_act(acc[tn][i] * scale + shift, act);
        if (res) v += res[o];
        out[o] = v;
      }
    }
  }
}

// ---------------------------------------------------------------------------
// Inference BN (elementwise over NCHW)
// ---------------------------------------------------------------------------
__global__ void k_bn(const float* __restrict__ in,
                     const float* g, const float* b, const float* m, const float* v,
                     float eps, float* __restrict__ out, long total, int C, long HW)
{
  long idx = (long)blockIdx.x * blockDim.x + threadIdx.x;
  if (idx >= total) return;
  int c = (int)((idx / HW) % C);
  float inv = rsqrtf(v[c] + eps) * g[c];
  out[idx] = in[idx] * inv + (b[c] - m[c] * inv);
}

// ---------------------------------------------------------------------------
// LayerNorm (one block per row)
// ---------------------------------------------------------------------------
__global__ __launch_bounds__(256)
void k_layernorm(const float* __restrict__ in, const float* __restrict__ g,
                 const float* __restrict__ b, float* __restrict__ out, int C)
{
  long row = blockIdx.x;
  const float* x = in + row * (long)C;
  float* o = out + row * (long)C;
  __shared__ float red[256];
  int tid = threadIdx.x;

  float s = 0.0f;
  for (int i = tid; i < C; i += 256) s += x[i];
  red[tid] = s; __syncthreads();
  for (int st = 128; st > 0; st >>= 1) {
    if (tid < st) red[tid] += red[tid + st];
    __syncthreads();
  }
  float mu = red[0] / (float)C;
  __syncthreads();

  float s2 = 0.0f;
  for (int i = tid; i < C; i += 256) { float d = x[i] - mu; s2 += d * d; }
  red[tid] = s2; __syncthreads();
  for (int st = 128; st > 0; st >>= 1) {
    if (tid < st) red[tid] += red[tid + st];
    __syncthreads();
  }
  float inv = rsqrtf(red[0] / (float)C + 1e-5f);
  for (int i = tid; i < C; i += 256)
    o[i] = (x[i] - mu) * inv * g[i] + b[i];
}

// ---------------------------------------------------------------------------
// Softmax over rows of length N, in place (one block per row)
// ---------------------------------------------------------------------------
__global__ __launch_bounds__(256)
void k_softmax(float* __restrict__ data, int N)
{
  long row = blockIdx.x;
  float* x = data + row * (long)N;
  __shared__ float red[256];
  int tid = threadIdx.x;

  float mx = -3.0e38f;
  for (int i = tid; i < N; i += 256) mx = fmaxf(mx, x[i]);
  red[tid] = mx; __syncthreads();
  for (int st = 128; st > 0; st >>= 1) {
    if (tid < st) red[tid] = fmaxf(red[tid], red[tid + st]);
    __syncthreads();
  }
  mx = red[0]; __syncthreads();

  float s = 0.0f;
  for (int i = tid; i < N; i += 256) { float e = expf(x[i] - mx); x[i] = e; s += e; }
  red[tid] = s; __syncthreads();
  for (int st = 128; st > 0; st >>= 1) {
    if (tid < st) red[tid] += red[tid + st];
    __syncthreads();
  }
  float inv = 1.0f / red[0];
  for (int i = tid; i < N; i += 256) x[i] *= inv;
}

// ---------------------------------------------------------------------------
// Generic batched transpose: out[b][j*ldo + colOff + i] = in[b][i*D1 + j]
// ---------------------------------------------------------------------------
__global__ void k_transpose(const float* __restrict__ in, float* __restrict__ out,
                            int Bn, int D0, int D1,
                            long inBatch, long outBatch, long ldo, int colOff)
{
  long total = (long)Bn * D0 * D1;
  long idx = (long)blockIdx.x * blockDim.x + threadIdx.x;
  if (idx >= total) return;
  int j = (int)(idx % D1); long t = idx / D1;
  int i = (int)(t % D0);   int b = (int)(t / D0);
  out[(long)b * outBatch + (long)j * ldo + colOff + i] =
      in[(long)b * inBatch + (long)i * D1 + j];
}

// ---------------------------------------------------------------------------
// Depthwise conv1d (k=4, pad (1,2)) + SiLU.
// ---------------------------------------------------------------------------
__global__ void k_dwconv4_silu(const float* __restrict__ x, long xBatch,
                               const float* __restrict__ w, float* __restrict__ out,
                               int Bn, int Dh, int L)
{
  long total = (long)Bn * Dh * L;
  long idx = (long)blockIdx.x * blockDim.x + threadIdx.x;
  if (idx >= total) return;
  int l = (int)(idx % L); long t = idx / L;
  int d = (int)(t % Dh);  int b = (int)(t / Dh);
  const float* xp = x + (long)b * xBatch + (long)d * L;
  const float* wp = w + (long)d * 4;
  float a = 0.0f;
#pragma unroll
  for (int k = 0; k < 4; ++k) {
    int li = l + k - 1;
    if (li >= 0 && li < L) a += xp[li] * wp[k];
  }
  out[idx] = a / (1.0f + expf(-a));
}

// delta = softplus(x + dtb[c])   (gemm already added dtb once -> net 2x dtb)
__global__ void k_softplus_bias(float* __restrict__ x, const float* __restrict__ bias,
                                long total, int C)
{
  long idx = (long)blockIdx.x * blockDim.x + threadIdx.x;
  if (idx >= total) return;
  int c = (int)(idx % C);
  x[idx] = softplusf(x[idx] + bias[c]);
}

// ---------------------------------------------------------------------------
// Selective scan: thread per (b, d). 8-state recurrence in registers.
// ---------------------------------------------------------------------------
__global__ void k_selective_scan(const float* __restrict__ u_t,
                                 const float* __restrict__ delta_t,
                                 const float* __restrict__ xd, int E, int r,
                                 const float* __restrict__ A_log,
                                 const float* __restrict__ Dp,
                                 float* __restrict__ y, int ldy,
                                 int Bn, int L, int Dh)
{
  int idx = blockIdx.x * blockDim.x + threadIdx.x;
  if (idx >= Bn * Dh) return;
  int d = idx % Dh, b = idx / Dh;

  float Av[8], s[8];
#pragma unroll
  for (int i = 0; i < 8; ++i) { Av[i] = -expf(A_log[(long)d * 8 + i]); s[i] = 0.0f; }
  float Dd = Dp[d];

  for (int l = 0; l < L; ++l) {
    long tok = (long)b * L + l;
    float de = delta_t[tok * Dh + d];
    float uu = u_t[tok * Dh + d];
    const float* xr = xd + tok * E + r;   // B at xr[0..7], C at xr[8..15]
    float yv = 0.0f;
#pragma unroll
    for (int i = 0; i < 8; ++i) {
      float dA = expf(de * Av[i]);
      s[i] = s[i] * dA + de * xr[i] * uu;
      yv += s[i] * xr[8 + i];
    }
    y[tok * ldy + d] = yv + Dd * uu;
  }
}

__global__ void k_add(const float* __restrict__ a, const float* __restrict__ b,
                      float* __restrict__ out, long n)
{
  long i = (long)blockIdx.x * blockDim.x + threadIdx.x;
  if (i < n) out[i] = a[i] + b[i];
}

__global__ void k_avgpool(const float* __restrict__ in, float* __restrict__ out,
                          int BC, int HW)
{
  int i = blockIdx.x * blockDim.x + threadIdx.x;
  if (i >= BC) return;
  const float* p = in + (long)i * HW;
  float s = 0.0f;
  for (int k = 0; k < HW; ++k) s += p[k];
  out[i] = s / (float)HW;
}

// ===========================================================================
// Host orchestration
// ===========================================================================
struct BNp  { const float *g, *b, *m, *v; };

static inline void gemm(hipStream_t s,
                        const float* A, long lda,
                        const float* B, long ldb, int bT,
                        const float* bias, float* C, long ldc,
                        int M, int N, int K, float alpha, int act,
                        int nz, BatchMap mA, BatchMap mB, BatchMap mC)
{
  dim3 grid((unsigned)((N + GEMM_BN - 1) / GEMM_BN),
            (unsigned)((M + GEMM_BM - 1) / GEMM_BM), (unsigned)nz);
  k_gemm_wmma<<<grid, 256, 0, s>>>(A, lda, B, ldb, bT, bias, C, ldc,
                                   M, N, K, alpha, act, mA, mB, mC);
}

extern "C" void kernel_launch(void* const* d_in, const int* in_sizes, int n_in,
                              void* d_out, int out_size, void* d_ws, size_t ws_size,
                              hipStream_t stream)
{
  (void)in_sizes; (void)n_in; (void)out_size; (void)ws_size;
  const BatchMap B1 = {1, 0, 0};

  // ---- parameter walk: setup_inputs() dict-insertion order, None dropped ----
  int ci = 0;
  auto nxt  = [&]() { return (const float*)d_in[ci++]; };
  auto rdBN = [&]() { BNp p; p.g = nxt(); p.b = nxt(); p.m = nxt(); p.v = nxt(); return p; };

  const float* x = nxt();                       // [16,3,224,224]
  const float* pe_w1 = nxt(); BNp pe_bn1 = rdBN();
  const float* pe_w2 = nxt(); BNp pe_bn2 = rdBN();

  struct ConvBlk { const float* w1; BNp bn1; const float* w2; BNp bn2; };
  struct MambaP  { const float *in_w, *in_b, *xp_w, *dtw, *dtb, *A_log, *Dp,
                               *out_w, *out_b, *cx, *cz; };
  struct AttnP   { const float *qkv_w, *proj_w, *proj_b; };
  struct MixBlk  { const float *ln1g, *ln1b, *ln2g, *ln2b;
                   int use_mamba; MambaP mp; AttnP ap;
                   const float *fc1w, *fc1b, *fc2w, *fc2b; };

  ConvBlk cb0[1], cb1[3];
  MixBlk  s2[8], s3[4];

  auto rdConv = [&](ConvBlk& c) {
    c.w1 = nxt(); c.bn1 = rdBN(); c.w2 = nxt(); c.bn2 = rdBN();
  };
  auto rdMix = [&](MixBlk& b, int mamba) {
    b.ln1g = nxt(); b.ln1b = nxt();
    b.ln2g = nxt(); b.ln2b = nxt();
    b.use_mamba = mamba;
    if (mamba) {
      MambaP& m = b.mp;
      m.in_w = nxt(); m.in_b = nxt();
      m.xp_w = nxt();                 // bias None -> dropped
      m.dtw  = nxt(); m.dtb = nxt();
      m.A_log = nxt(); m.Dp = nxt();
      m.out_w = nxt(); m.out_b = nxt();
      m.cx = nxt(); m.cz = nxt();
    } else {
      AttnP& a = b.ap;
      a.qkv_w = nxt();                // bias None -> dropped
      a.proj_w = nxt(); a.proj_b = nxt();
    }
    b.fc1w = nxt(); b.fc1b = nxt();
    b.fc2w = nxt(); b.fc2b = nxt();
  };

  rdConv(cb0[0]);
  for (int j = 0; j < 3; ++j) rdConv(cb1[j]);
  for (int j = 0; j < 8; ++j) rdMix(s2[j], j < 4);   // depth 8 -> 4 mamba
  for (int j = 0; j < 4; ++j) rdMix(s3[j], j < 2);   // depth 4 -> 2 mamba
  const float* down_w[3]; for (int i = 0; i < 3; ++i) down_w[i] = nxt();
  BNp norm = rdBN();
  const float* head_w = nxt(); const float* head_b = nxt();

  // ---- workspace arena (floats; all offsets keep 16B alignment) ----
  float* P = (float*)d_ws;
  size_t off = 0;
  auto alloc = [&](size_t n) { float* p = P + off; off += n; return p; };
  float* actA = alloc(6600000);   // NCHW ping (max 16*32*112*112 = 6.43M)
  float* actB = alloc(6600000);   // NCHW pong
  float* actC = alloc(4300000);   // NCHW third (conv-block residual dance)
  float* tkn  = alloc(1100000);   // residual token stream [tok, C]
  float* hbuf = alloc(1100000);   // post-LN tokens
  float* big0 = alloc(5100000);   // xz / qkv / MLP hidden
  float* big1 = alloc(5100000);   // xz^T / attention scores
  float* sm0  = alloc(1100000);   // x after dwconv  [B,dh,L]
  float* sm1  = alloc(1100000);   // z after dwconv  [B,dh,L]
  float* sm2  = alloc(1100000);   // x_t token-major [tok, dh]
  float* ycat = alloc(1100000);   // [y | z] concat  [tok, C]
  float* xdb  = alloc(300000);    // xd projection   [tok, E]
  float* delt = alloc(1100000);   // delta           [tok, dh]
  float* mixb = alloc(1400000);   // mixer/MLP output before residual

  auto conv = [&](const float* in, const float* w, const BNp* bn, float eps,
                  const float* res, float* out,
                  int Nb, int Ci, int Hi, int Wi, int Co, int stride, int act) {
    int Ho = (Hi - 1) / stride + 1, Wo = (Wi - 1) / stride + 1;
    int M = Nb * Ho * Wo;
    dim3 grid((unsigned)((Co + GEMM_BN - 1) / GEMM_BN),
              (unsigned)((M + GEMM_BM - 1) / GEMM_BM), 1);
    k_conv3x3_wmma<<<grid, 256, 0, stream>>>(
        in, w, bn ? bn->g : nullptr, bn ? bn->b : nullptr,
        bn ? bn->m : nullptr, bn ? bn->v : nullptr, eps, res, out,
        Nb, Ci, Hi, Wi, Co, stride, act);
  };
  auto transpose = [&](const float* in, float* out, int Bn, int D0, int D1,
                       long ldo, long outBatch, int colOff) {
    long total = (long)Bn * D0 * D1;
    k_transpose<<<(unsigned)((total + 255) / 256), 256, 0, stream>>>(
        in, out, Bn, D0, D1, (long)D0 * D1, outBatch, ldo, colOff);
  };
  auto addinto = [&](float* dst, const float* add, long n) {
    k_add<<<(unsigned)((n + 255) / 256), 256, 0, stream>>>(dst, add, dst, n);
  };

  // ---- token-mixer stage (ws == H == W, so partition == transpose) ----
  auto run_mixer_stage = [&](MixBlk* blks, int nblk, int Bn, int L, int Cd, int heads) {
    const int tok = Bn * L;
    const int dh = Cd / 2, r = (Cd + 15) / 16, E = r + 16, n3 = 3 * Cd;
    const int hd = Cd / heads;
    const float scale = 1.0f / sqrtf((float)hd);
    for (int j = 0; j < nblk; ++j) {
      MixBlk& p = blks[j];
      k_layernorm<<<(unsigned)tok, 256, 0, stream>>>(tkn, p.ln1g, p.ln1b, hbuf, Cd);
      if (p.use_mamba) {
        // xz = h @ in_w^T + in_b
        gemm(stream, hbuf, Cd, p.mp.in_w, Cd, 1, p.mp.in_b, big0, Cd,
             tok, Cd, Cd, 1.0f, ACT_NONE, 1, B1, B1, B1);
        // [B,L,C] -> [B,C,L]
        transpose(big0, big1, Bn, L, Cd, (long)L, (long)Cd * L, 0);
        long xb = (long)Cd * L;
        long tot = (long)Bn * dh * L;
        k_dwconv4_silu<<<(unsigned)((tot + 255) / 256), 256, 0, stream>>>(
            big1, xb, p.mp.cx, sm0, Bn, dh, L);
        k_dwconv4_silu<<<(unsigned)((tot + 255) / 256), 256, 0, stream>>>(
            big1 + (long)dh * L, xb, p.mp.cz, sm1, Bn, dh, L);
        // x_t: [B,dh,L] -> [tok, dh]
        transpose(sm0, sm2, Bn, dh, L, (long)dh, (long)L * dh, 0);
        // xd = x_t @ xp_w^T : [tok, E]
        gemm(stream, sm2, dh, p.mp.xp_w, dh, 1, nullptr, xdb, E,
             tok, E, dh, 1.0f, ACT_NONE, 1, B1, B1, B1);
        // dt = xd[:, :r] @ dtw^T + dtb ; delta = softplus(dt + dtb)
        gemm(stream, xdb, E, p.mp.dtw, r, 1, p.mp.dtb, delt, dh,
             tok, dh, r, 1.0f, ACT_NONE, 1, B1, B1, B1);
        k_softplus_bias<<<(unsigned)(((long)tok * dh + 255) / 256), 256, 0, stream>>>(
            delt, p.mp.dtb, (long)tok * dh, dh);
        // scan -> ycat[:, :dh]
        k_selective_scan<<<(unsigned)((Bn * dh + 255) / 256), 256, 0, stream>>>(
            sm2, delt, xdb, E, r, p.mp.A_log, p.mp.Dp, ycat, Cd, Bn, L, dh);
        // z_t -> ycat[:, dh:]
        transpose(sm1, ycat, Bn, dh, L, (long)Cd, (long)L * Cd, dh);
        // out projection
        gemm(stream, ycat, Cd, p.mp.out_w, Cd, 1, p.mp.out_b, mixb, Cd,
             tok, Cd, Cd, 1.0f, ACT_NONE, 1, B1, B1, B1);
      } else {
        // qkv = h @ qkv_w^T : [tok, 3C]
        gemm(stream, hbuf, Cd, p.ap.qkv_w, Cd, 1, nullptr, big0, n3,
             tok, n3, Cd, 1.0f, ACT_NONE, 1, B1, B1, B1);
        // scores = scale * Q K^T, batched over (window, head)
        BatchMap mQ = {heads, (long)L * n3, (long)hd};
        BatchMap mS = {1, (long)L * L, 0};
        gemm(stream, big0, n3, big0 + Cd, n3, 1, nullptr, big1, L,
             L, L, hd, scale, ACT_NONE, Bn * heads, mQ, mQ, mS);
        k_softmax<<<(unsigned)(Bn * heads * L), 256, 0, stream>>>(big1, L);
        // out = scores @ V  -> ycat[tok, C] at head offsets
        BatchMap mO = {heads, (long)L * Cd, (long)hd};
        gemm(stream, big1, L, big0 + 2 * Cd, n3, 0, nullptr, ycat, Cd,
             L, hd, L, 1.0f, ACT_NONE, Bn * heads, mS, mQ, mO);
        gemm(stream, ycat, Cd, p.ap.proj_w, Cd, 1, p.ap.proj_b, mixb, Cd,
             tok, Cd, Cd, 1.0f, ACT_NONE, 1, B1, B1, B1);
      }
      addinto(tkn, mixb, (long)tok * Cd);
      // MLP
      k_layernorm<<<(unsigned)tok, 256, 0, stream>>>(tkn, p.ln2g, p.ln2b, hbuf, Cd);
      gemm(stream, hbuf, Cd, p.fc1w, Cd, 1, p.fc1b, big0, 4 * Cd,
           tok, 4 * Cd, Cd, 1.0f, ACT_GELU_ERF, 1, B1, B1, B1);
      gemm(stream, big0, 4 * Cd, p.fc2w, 4 * Cd, 1, p.fc2b, mixb, Cd,
           tok, Cd, 4 * Cd, 1.0f, ACT_NONE, 1, B1, B1, B1);
      addinto(tkn, mixb, (long)tok * Cd);
    }
  };

  // ================= forward =================
  // patch embed: 3->32 s2 (bn eps 1e-4, relu), 32->80 s2
  conv(x, pe_w1, &pe_bn1, 1e-4f, nullptr, actA, 16, 3, 224, 224, 32, 2, ACT_RELU);
  conv(actA, pe_w2, &pe_bn2, 1e-4f, nullptr, actB, 16, 32, 112, 112, 80, 2, ACT_RELU);
  float* cur = actB;
  auto pick2 = [&](float*& o1, float*& o2) {
    if (cur == actA)      { o1 = actB; o2 = actC; }
    else if (cur == actB) { o1 = actA; o2 = actC; }
    else                  { o1 = actA; o2 = actB; }
  };

  // stage 0: 1 conv block @ 56x56, d=80
  {
    float *o1, *o2; pick2(o1, o2);
    conv(cur, cb0[0].w1, &cb0[0].bn1, 1e-5f, nullptr, o1, 16, 80, 56, 56, 80, 1, ACT_GELU_TANH);
    conv(o1, cb0[0].w2, &cb0[0].bn2, 1e-5f, cur, o2, 16, 80, 56, 56, 80, 1, ACT_NONE);
    cur = o2;
  }
  { float *o1, *o2; pick2(o1, o2);   // down0: 80->160 s2 -> 28x28
    conv(cur, down_w[0], nullptr, 0.f, nullptr, o1, 16, 80, 56, 56, 160, 2, ACT_NONE);
    cur = o1; }

  // stage 1: 3 conv blocks @ 28x28, d=160
  for (int j = 0; j < 3; ++j) {
    float *o1, *o2; pick2(o1, o2);
    conv(cur, cb1[j].w1, &cb1[j].bn1, 1e-5f, nullptr, o1, 16, 160, 28, 28, 160, 1, ACT_GELU_TANH);
    conv(o1, cb1[j].w2, &cb1[j].bn2, 1e-5f, cur, o2, 16, 160, 28, 28, 160, 1, ACT_NONE);
    cur = o2;
  }
  { float *o1, *o2; pick2(o1, o2);   // down1: 160->320 s2 -> 14x14
    conv(cur, down_w[1], nullptr, 0.f, nullptr, o1, 16, 160, 28, 28, 320, 2, ACT_NONE);
    cur = o1; }

  // stage 2: 14x14, ws=14 -> partition == NCHW->NLC transpose; L=196, C=320
  transpose(cur, tkn, 16, 320, 196, 320, (long)196 * 320, 0);
  run_mixer_stage(s2, 8, 16, 196, 320, 8);
  { float *o1, *o2; pick2(o1, o2);
    transpose(tkn, o1, 16, 196, 320, 196, (long)320 * 196, 0);
    cur = o1; }
  { float *o1, *o2; pick2(o1, o2);   // down2: 320->640 s2 -> 7x7
    conv(cur, down_w[2], nullptr, 0.f, nullptr, o1, 16, 320, 14, 14, 640, 2, ACT_NONE);
    cur = o1; }

  // stage 3: 7x7, ws=7; L=49, C=640
  transpose(cur, tkn, 16, 640, 49, 640, (long)49 * 640, 0);
  run_mixer_stage(s3, 4, 16, 49, 640, 16);
  { float *o1, *o2; pick2(o1, o2);
    transpose(tkn, o1, 16, 49, 640, 49, (long)640 * 49, 0);
    cur = o1; }

  // final norm + global avg pool + head
  { float *o1, *o2; pick2(o1, o2);
    long tot = (long)16 * 640 * 49;
    k_bn<<<(unsigned)((tot + 255) / 256), 256, 0, stream>>>(
        cur, norm.g, norm.b, norm.m, norm.v, 1e-5f, o1, tot, 640, 49);
    cur = o1; }
  k_avgpool<<<(unsigned)((16 * 640 + 255) / 256), 256, 0, stream>>>(cur, sm0, 16 * 640, 49);
  gemm(stream, sm0, 640, head_w, 640, 1, head_b, (float*)d_out, 100,
       16, 100, 640, 1.0f, ACT_NONE, 1, B1, B1, B1);
}
